// FullCausalAttention_81466939670766
// MI455X (gfx1250) — matrix operand order repaired and checked by reading, not compiled, not run on hardware
//
#include <hip/hip_runtime.h>
#include <hip/hip_bf16.h>

// ---------------- problem constants ----------------
#define BATCH 2
#define SEQ   2048
#define DIM   2048
#define NH    16
#define KVH   4
#define HD    128
#define GROUPS (NH / KVH)   // 4
#define QKDIM (NH * HD)     // 2048
#define KVDIM (KVH * HD)    // 512
#define MTOK  (BATCH * SEQ) // 4096 tokens

typedef __attribute__((ext_vector_type(8)))  float  v8f;
typedef __attribute__((ext_vector_type(16))) __bf16 v16bf;
typedef __attribute__((ext_vector_type(8)))  __bf16 v8bf;

static __device__ __forceinline__ v8bf ld8(const __bf16* p) { return *(const v8bf*)p; }

static __device__ __forceinline__ v16bf vcat(v8bf lo, v8bf hi) {
    v16bf r;
#pragma unroll
    for (int i = 0; i < 8; ++i) { r[i] = lo[i]; r[i + 8] = hi[i]; }
    return r;
}

static __device__ __forceinline__ v8f wmma_bf16(v16bf a, v16bf b, v8f c) {
    // D = A(16x32 bf16) * B(32x16 bf16) + C(16x16 f32)
    return __builtin_amdgcn_wmma_f32_16x16x32_bf16(
        /*neg_a=*/false, a, /*neg_b=*/false, b,
        /*c_mod=*/(short)0, c, /*reuse_a=*/false, /*reuse_b=*/false);
}

// Async copy 16B: global -> LDS, tracked by ASYNCcnt (CDNA5 §08).
static __device__ __forceinline__ void async_g2l_b128(unsigned lds_byte_addr,
                                                      const void* gaddr) {
    unsigned long long ga = (unsigned long long)(uintptr_t)gaddr;
    asm volatile("global_load_async_to_lds_b128 %0, %1, off"
                 :: "v"(lds_byte_addr), "v"(ga) : "memory");
}
static __device__ __forceinline__ void wait_asynccnt0() {
    asm volatile("s_wait_asynccnt 0x0" ::: "memory");
}
// Low 32 bits of a flat pointer to LDS == LDS byte address (aperture rule).
static __device__ __forceinline__ unsigned lds_addr_of(const void* p) {
    return (unsigned)(uintptr_t)p;
}

// ---------------- elementwise converters ----------------
__global__ __launch_bounds__(256) void cvt_f32_bf16(const float* __restrict__ in,
                                                    __bf16* __restrict__ out, int n) {
    int i = blockIdx.x * blockDim.x + threadIdx.x;
    if (i < n) out[i] = (__bf16)in[i];
}

// out[c][r] = (bf16) in[r][c]   (in: R rows x C cols, row-major)
__global__ __launch_bounds__(256) void transpose_cvt(const float* __restrict__ in,
                                                     __bf16* __restrict__ out,
                                                     int R, int C) {
    int i = blockIdx.x * blockDim.x + threadIdx.x;
    if (i >= R * C) return;
    int r = i / C, c = i % C;
    out[(size_t)c * R + r] = (__bf16)in[(size_t)r * C + c];
}

// Vt[b][kv][d][n] = V[b][n][kv][d]
__global__ __launch_bounds__(256) void transpose_v(const __bf16* __restrict__ V,
                                                   __bf16* __restrict__ Vt) {
    int i = blockIdx.x * blockDim.x + threadIdx.x;
    const int total = BATCH * KVH * HD * SEQ;
    if (i >= total) return;
    int n = i % SEQ;
    int d = (i / SEQ) % HD;
    int kv = (i / (SEQ * HD)) % KVH;
    int b = i / (SEQ * HD * KVH);
    Vt[i] = V[(((size_t)b * SEQ + n) * KVH + kv) * HD + d];
}

// ---------------- RoPE (in place, bf16 storage, f32 math) ----------------
__global__ __launch_bounds__(256) void rope_kernel(__bf16* __restrict__ t, int nheads) {
    int i = blockIdx.x * blockDim.x + threadIdx.x;
    const int total = BATCH * SEQ * nheads * (HD / 2);
    if (i >= total) return;
    int d = i % (HD / 2);
    int h = (i / (HD / 2)) % nheads;
    int n = (i / ((HD / 2) * nheads)) % SEQ;
    int b = i / ((HD / 2) * nheads * SEQ);
    __bf16* p = t + (((size_t)b * SEQ + n) * nheads + h) * HD;
    float x1 = (float)p[d];
    float x2 = (float)p[d + HD / 2];
    float inv_freq = __powf(10000.0f, -(2.0f * (float)d) / (float)HD);
    float ang = (float)n * inv_freq;
    float s, c;
    __sincosf(ang, &s, &c);
    p[d]          = (__bf16)(x1 * c - x2 * s);
    p[d + HD / 2] = (__bf16)(x2 * c + x1 * s);
}

// ---------------- bf16 WMMA GEMM: C[M][Nn] = A[M][K] * BT[Nn][K]^T --------
// 4 waves/block; wave = 16(M) x 64(N) strip; register double-buffered K-loop
// so global loads for tile k+32 overlap the 4 WMMAs of tile k.
__global__ __launch_bounds__(128) void gemm_bf16(const __bf16* __restrict__ A,
                                                 const __bf16* __restrict__ BT,
                                                 __bf16* __restrict__ Cb,
                                                 float* __restrict__ Cf,
                                                 int M, int Nn, int K) {
    const int lane = threadIdx.x & 31;
    const int wave = threadIdx.x >> 5;
    const int half = lane >> 4;
    const int l15  = lane & 15;
    const int m0 = blockIdx.x * 16;
    const int n0 = blockIdx.y * 256 + wave * 64;

    v8f acc0 = (v8f){0,0,0,0,0,0,0,0}, acc1 = acc0, acc2 = acc0, acc3 = acc0;

    const __bf16* arow = A + (size_t)(m0 + l15) * K;
    const __bf16* brow0 = BT + (size_t)(n0 +  0 + l15) * K + half * 16;
    const __bf16* brow1 = BT + (size_t)(n0 + 16 + l15) * K + half * 16;
    const __bf16* brow2 = BT + (size_t)(n0 + 32 + l15) * K + half * 16;
    const __bf16* brow3 = BT + (size_t)(n0 + 48 + l15) * K + half * 16;
    const int ka0 = half * 16;
    const int ka1 = half * 16 + 8;

    v16bf av = vcat(ld8(arow + ka0), ld8(arow + ka1));
    v16bf b0 = vcat(ld8(brow0), ld8(brow0 + 8));
    v16bf b1 = vcat(ld8(brow1), ld8(brow1 + 8));
    v16bf b2 = vcat(ld8(brow2), ld8(brow2 + 8));
    v16bf b3 = vcat(ld8(brow3), ld8(brow3 + 8));

    for (int k = 0; k < K - 32; k += 32) {
        const int kn = k + 32;
        __builtin_prefetch(arow + kn + 32, 0, 0);
        v16bf avn = vcat(ld8(arow + kn + ka0), ld8(arow + kn + ka1));
        v16bf c0 = vcat(ld8(brow0 + kn), ld8(brow0 + kn + 8));
        v16bf c1 = vcat(ld8(brow1 + kn), ld8(brow1 + kn + 8));
        v16bf c2 = vcat(ld8(brow2 + kn), ld8(brow2 + kn + 8));
        v16bf c3 = vcat(ld8(brow3 + kn), ld8(brow3 + kn + 8));
        acc0 = wmma_bf16(av, b0, acc0);
        acc1 = wmma_bf16(av, b1, acc1);
        acc2 = wmma_bf16(av, b2, acc2);
        acc3 = wmma_bf16(av, b3, acc3);
        av = avn; b0 = c0; b1 = c1; b2 = c2; b3 = c3;
    }
    acc0 = wmma_bf16(av, b0, acc0);
    acc1 = wmma_bf16(av, b1, acc1);
    acc2 = wmma_bf16(av, b2, acc2);
    acc3 = wmma_bf16(av, b3, acc3);

    v8f accs[4] = {acc0, acc1, acc2, acc3};
#pragma unroll
    for (int t = 0; t < 4; ++t) {
#pragma unroll
        for (int r = 0; r < 8; ++r) {
            int row = m0 + r + half * 8;
            int col = n0 + t * 16 + l15;
            if (Cf) Cf[(size_t)row * Nn + col] = accs[t][r];
            else    Cb[(size_t)row * Nn + col] = (__bf16)accs[t][r];
        }
    }
}

// ---------------- flash attention (causal, GQA) ----------------
// grid (SEQ/64, NH, BATCH), block 128 (4 waves). K/V tiles are staged into LDS
// with double-buffered async copies (ASYNCcnt) shared by all 4 waves.
// All waves run a uniform number of key tiles; fully-masked tiles are numeric
// no-ops in the online softmax, keeping in-loop barriers legal.
__global__ __launch_bounds__(128) void attn_kernel(const __bf16* __restrict__ Q,  // [B][N][NH][HD]
                                                   const __bf16* __restrict__ Kk, // [B][N][KVH][HD]
                                                   const __bf16* __restrict__ Vt, // [B][KVH][HD][N]
                                                   __bf16* __restrict__ Ao) {     // [B][N][NH*HD]
    __shared__ __bf16 kbuf[2][32][HD];   // keys x d       (8 KB per buffer)
    __shared__ __bf16 vbuf[2][HD][32];   // d x keys       (8 KB per buffer)
    __shared__ __bf16 plds[4][16 * 32];  // per-wave P staging

    const int lane = threadIdx.x & 31;
    const int wave = threadIdx.x >> 5;
    const int half = lane >> 4;
    const int l15  = lane & 15;
    const int tid  = threadIdx.x;
    const int b = blockIdx.z;
    const int h = blockIdx.y;
    const int kv = h / GROUPS;
    const int q0 = (blockIdx.x * 4 + wave) * 16;
    const float scale = 0.08838834764831845f;   // 128^-0.5

    const __bf16* Kbase = Kk + ((size_t)b * SEQ * KVH + kv) * HD;
    const __bf16* Vbase = Vt + ((size_t)b * KVH + kv) * HD * SEQ;

    // cooperative async stage of key-tile j into buffer buf
    auto issue_tile = [&](int j, int buf) {
        const int kbase = j * 32;
        const __bf16* Kg = Kbase + (size_t)kbase * (KVH * HD);
        const __bf16* Vg = Vbase + kbase;
        unsigned kl = lds_addr_of(&kbuf[buf][0][0]);
        unsigned vl = lds_addr_of(&vbuf[buf][0][0]);
#pragma unroll
        for (int c = tid; c < 512; c += 128) {
            async_g2l_b128(kl + c * 16, Kg + (size_t)(c >> 4) * (KVH * HD) + (c & 15) * 8);
            async_g2l_b128(vl + c * 16, Vg + (size_t)(c >> 2) * SEQ + (c & 3) * 8);
        }
    };

    // Q as A-operands: 4 chunks of Kdim=32 over d (held in registers)
    const __bf16* qrow = Q + (((size_t)b * SEQ + q0 + l15) * NH + h) * HD;
    v16bf qa[4];
#pragma unroll
    for (int dc = 0; dc < 4; ++dc)
        qa[dc] = vcat(ld8(qrow + dc * 32 + half * 16),
                      ld8(qrow + dc * 32 + half * 16 + 8));

    v8f o[8];
#pragma unroll
    for (int dt = 0; dt < 8; ++dt) o[dt] = (v8f){0,0,0,0,0,0,0,0};
    float mrow[8], lrow[8];
#pragma unroll
    for (int r = 0; r < 8; ++r) { mrow[r] = -3.0e38f; lrow[r] = 0.f; }

    __bf16* pl = plds[wave];
    const int jmax = blockIdx.x * 2 + 2;        // uniform across the block

    issue_tile(0, 0);
    wait_asynccnt0();
    __syncthreads();

    for (int j = 0; j < jmax; ++j) {
        const int cur = j & 1;
        const int kbase = j * 32;
        if (j + 1 < jmax) issue_tile(j + 1, cur ^ 1);

        // ---- S = Q K^T from LDS (two 16x16 tiles over 32 keys) ----
        v8f s[2];
#pragma unroll
        for (int t = 0; t < 2; ++t) {
            v8f acc = (v8f){0,0,0,0,0,0,0,0};
            const __bf16* krow = &kbuf[cur][t * 16 + l15][half * 16];
#pragma unroll
            for (int dc = 0; dc < 4; ++dc) {
                v16bf bv = vcat(ld8(krow + dc * 32), ld8(krow + dc * 32 + 8));
                acc = wmma_bf16(qa[dc], bv, acc);
            }
            s[t] = acc;
        }

        // ---- scale + causal mask + online softmax ----
        float corr[8];
#pragma unroll
        for (int r = 0; r < 8; ++r) {
            const int qidx = q0 + r + half * 8;
#pragma unroll
            for (int t = 0; t < 2; ++t) {
                const int key = kbase + t * 16 + l15;
                float val = s[t][r] * scale;
                s[t][r] = (key <= qidx) ? val : -3.0e38f;
            }
            float mx = fmaxf(s[0][r], s[1][r]);
#pragma unroll
            for (int off = 1; off < 16; off <<= 1)     // stays inside 16-lane half
                mx = fmaxf(mx, __shfl_xor(mx, off, 32));
            float mnew = fmaxf(mrow[r], mx);
            corr[r] = __expf(mrow[r] - mnew);
            mrow[r] = mnew;
            float p0 = __expf(s[0][r] - mnew);
            float p1 = __expf(s[1][r] - mnew);
            s[0][r] = p0; s[1][r] = p1;
            float sum = p0 + p1;
#pragma unroll
            for (int off = 1; off < 16; off <<= 1)
                sum += __shfl_xor(sum, off, 32);
            lrow[r] = lrow[r] * corr[r] + sum;
        }
#pragma unroll
        for (int dt = 0; dt < 8; ++dt)
#pragma unroll
            for (int r = 0; r < 8; ++r) o[dt][r] *= corr[r];

        // ---- P: C layout -> LDS -> A layout (DS ops in-order per wave) ----
#pragma unroll
        for (int t = 0; t < 2; ++t)
#pragma unroll
            for (int r = 0; r < 8; ++r)
                pl[(r + half * 8) * 32 + t * 16 + l15] = (__bf16)s[t][r];
        asm volatile("s_wait_dscnt 0x0" ::: "memory");
        v16bf pa = vcat(ld8(pl + l15 * 32 + half * 16),
                        ld8(pl + l15 * 32 + half * 16 + 8));

        // ---- O += P V from LDS (8 d-tiles of 16) ----
#pragma unroll
        for (int dt = 0; dt < 8; ++dt) {
            const __bf16* vrow = &vbuf[cur][dt * 16 + l15][half * 16];
            v16bf bv = vcat(ld8(vrow), ld8(vrow + 8));
            o[dt] = wmma_bf16(pa, bv, o[dt]);
        }

        wait_asynccnt0();      // tile j+1 landed
        __syncthreads();       // visible to all waves; buffers safe to reuse
    }

    // ---- finalize + store bf16 [B][N][NH*HD] ----
    float inv[8];
#pragma unroll
    for (int r = 0; r < 8; ++r) inv[r] = 1.0f / lrow[r];
#pragma unroll
    for (int dt = 0; dt < 8; ++dt)
#pragma unroll
        for (int r = 0; r < 8; ++r) {
            int row = q0 + r + half * 8;
            Ao[((size_t)b * SEQ + row) * QKDIM + h * HD + dt * 16 + l15] =
                (__bf16)(o[dt][r] * inv[r]);
        }
}

// ---------------- host orchestration ----------------
extern "C" void kernel_launch(void* const* d_in, const int* in_sizes, int n_in,
                              void* d_out, int out_size, void* d_ws, size_t ws_size,
                              hipStream_t stream) {
    const float* x  = (const float*)d_in[0];
    const float* Wq = (const float*)d_in[1];
    const float* Wk = (const float*)d_in[2];
    const float* Wv = (const float*)d_in[3];
    const float* Wo = (const float*)d_in[4];
    float* out = (float*)d_out;

    char* ws = (char*)d_ws;
    size_t off = 0;
    auto alloc = [&](size_t bytes) { char* p = ws + off; off += (bytes + 255) & ~size_t(255); return p; };

    __bf16* xb  = (__bf16*)alloc((size_t)MTOK * DIM * 2);
    __bf16* WqT = (__bf16*)alloc((size_t)QKDIM * DIM * 2);
    __bf16* WkT = (__bf16*)alloc((size_t)KVDIM * DIM * 2);
    __bf16* WvT = (__bf16*)alloc((size_t)KVDIM * DIM * 2);
    __bf16* WoT = (__bf16*)alloc((size_t)DIM * QKDIM * 2);
    __bf16* Qb  = (__bf16*)alloc((size_t)MTOK * QKDIM * 2);
    __bf16* Kb  = (__bf16*)alloc((size_t)MTOK * KVDIM * 2);
    __bf16* Vb  = (__bf16*)alloc((size_t)MTOK * KVDIM * 2);
    __bf16* Vt  = (__bf16*)alloc((size_t)MTOK * KVDIM * 2);
    __bf16* Ao  = (__bf16*)alloc((size_t)MTOK * QKDIM * 2);

    {
        int n = MTOK * DIM;
        cvt_f32_bf16<<<(n + 255) / 256, 256, 0, stream>>>(x, xb, n);
    }
    transpose_cvt<<<(DIM * QKDIM + 255) / 256, 256, 0, stream>>>(Wq, WqT, DIM, QKDIM);
    transpose_cvt<<<(DIM * KVDIM + 255) / 256, 256, 0, stream>>>(Wk, WkT, DIM, KVDIM);
    transpose_cvt<<<(DIM * KVDIM + 255) / 256, 256, 0, stream>>>(Wv, WvT, DIM, KVDIM);
    transpose_cvt<<<(QKDIM * DIM + 255) / 256, 256, 0, stream>>>(Wo, WoT, QKDIM, DIM);

    gemm_bf16<<<dim3(MTOK / 16, QKDIM / 256), 128, 0, stream>>>(xb, WqT, Qb, nullptr, MTOK, QKDIM, DIM);
    gemm_bf16<<<dim3(MTOK / 16, KVDIM / 256), 128, 0, stream>>>(xb, WkT, Kb, nullptr, MTOK, KVDIM, DIM);
    gemm_bf16<<<dim3(MTOK / 16, KVDIM / 256), 128, 0, stream>>>(xb, WvT, Vb, nullptr, MTOK, KVDIM, DIM);

    {
        int nq = BATCH * SEQ * NH * (HD / 2);
        rope_kernel<<<(nq + 255) / 256, 256, 0, stream>>>(Qb, NH);
        int nk = BATCH * SEQ * KVH * (HD / 2);
        rope_kernel<<<(nk + 255) / 256, 256, 0, stream>>>(Kb, KVH);
    }

    {
        int n = BATCH * KVH * HD * SEQ;
        transpose_v<<<(n + 255) / 256, 256, 0, stream>>>(Vb, Vt);
    }

    attn_kernel<<<dim3(SEQ / 64, NH, BATCH), 128, 0, stream>>>(Qb, Kb, Vt, Ao);

    gemm_bf16<<<dim3(MTOK / 16, DIM / 256), 128, 0, stream>>>(Ao, WoT, nullptr, out, MTOK, DIM, DIM);
}
